// GraphAttentionLayer_23347442221666
// MI455X (gfx1250) — matrix operand order repaired and checked by reading, not compiled, not run on hardware
//
#include <hip/hip_runtime.h>

// ---------- types ----------
typedef __attribute__((ext_vector_type(16))) __bf16  v16bf;
typedef __attribute__((ext_vector_type(8)))  float   v8f;
typedef __attribute__((ext_vector_type(4)))  unsigned int v4u;
typedef __attribute__((ext_vector_type(4)))  int     v4i;
typedef __attribute__((ext_vector_type(8)))  int     v8i;

union Frag16 { v4u u[2]; v16bf b; };

#define BATCH 16
#define NN    1024
#define DD    256
#define NEG_SLOPE 0.2f
#define MASKVAL  (-1.0e30f)

// round-to-nearest-even f32 -> bf16
static __device__ __forceinline__ unsigned short f2bf(float f) {
    unsigned int u = __float_as_uint(f);
    unsigned int r = u + 0x7FFFu + ((u >> 16) & 1u);
    return (unsigned short)(r >> 16);
}
static __device__ __forceinline__ unsigned int packbf(float lo, float hi) {
    return (unsigned int)f2bf(lo) | ((unsigned int)f2bf(hi) << 16);
}

// =====================================================================
// Kernel 0: swizzle W (DOUT x DIN, f32 row-major) into bf16 B-fragment
// layout: [kt(8)][otile(16)][lane(32)][v(8)][h(2)]  (32B contiguous/lane)
// element(lane l, v, h) = W[otile*16 + (l&15)][kt*32 + 2v + 16*(l>>4) + h]
// =====================================================================
__global__ __launch_bounds__(256) void k_wswz(const float* __restrict__ W,
                                              unsigned short* __restrict__ WbfB) {
    int t  = blockIdx.x * blockDim.x + threadIdx.x;   // 65536 threads
    int h_ = t & 1;
    int v  = (t >> 1) & 7;
    int l  = (t >> 4) & 31;
    int ot = (t >> 9) & 15;
    int kt = t >> 13;
    int col = ot * 16 + (l & 15);
    int k   = kt * 32 + 2 * v + 16 * (l >> 4) + h_;
    WbfB[t] = f2bf(W[col * DD + k]);
}

// =====================================================================
// Kernel 1: Wh = h @ W^T  (bf16 WMMA, f32 accum), emit:
//   * Wh in swizzled bf16 B-fragment layout for the 2nd GEMM
//   * ei = Wh . a1 , ej = Wh . a2   (f32, deterministic in-WG reduction)
// grid = 1024 row-tiles (16 rows each), block = 512 (16 waves, 1/otile)
// =====================================================================
__global__ __launch_bounds__(512) void k_wh(const float* __restrict__ h,
                                            const float* __restrict__ avec,
                                            const unsigned short* __restrict__ WbfB,
                                            unsigned short* __restrict__ WhB,
                                            float* __restrict__ ei,
                                            float* __restrict__ ej) {
    __shared__ unsigned int sA[8 * 256];    // 8KB: 8 K-tiles x 1024B A-fragment blocks
    __shared__ float sEi[16][16];
    __shared__ float sEj[16][16];

    const int rt   = blockIdx.x;       // row tile 0..1023
    const int r0   = rt * 16;
    const int tid  = threadIdx.x;
    const int wave = tid >> 5;         // = otile
    const int L    = tid & 31;
    const int col  = L & 15;
    const int kh   = L >> 4;

    // ---- cooperative load of h tile into LDS, pre-swizzled to A-fragment ----
    {
        const int i  = tid >> 5;       // row 0..15
        const int c  = tid & 31;       // k-octet 0..31
        const float4* hp = (const float4*)(h + (size_t)(r0 + i) * DD + c * 8);
        float4 x0 = hp[0];
        float4 x1 = hp[1];
        unsigned int p0 = packbf(x0.x, x0.y);
        unsigned int p1 = packbf(x0.z, x0.w);
        unsigned int p2 = packbf(x1.x, x1.y);
        unsigned int p3 = packbf(x1.z, x1.w);
        const int kt  = c >> 2;
        const int kk0 = (c & 3) * 8;
        const int q   = (kk0 >> 4) & 1;
        const int v0  = ((kk0 >> 3) & 1) * 4;
        v4u st = {p0, p1, p2, p3};
        ((v4u*)sA)[(kt * 256 + (i + 16 * q) * 8 + v0) >> 2] = st;
    }
    __syncthreads();

    // ---- 8x WMMA over K = 256 ----
    v8f acc = {};
    const int otile = wave;
    #pragma unroll
    for (int kt = 0; kt < 8; ++kt) {
        Frag16 fa, fb;
        const v4u* ap = (const v4u*)(sA + kt * 256 + L * 8);
        fa.u[0] = ap[0];
        fa.u[1] = ap[1];
        const v4u* bp = (const v4u*)(WbfB + (size_t)(kt * 16 + otile) * 512 + L * 16);
        fb.u[0] = bp[0];
        fb.u[1] = bp[1];
        acc = __builtin_amdgcn_wmma_f32_16x16x32_bf16(
            false, fa.b, false, fb.b, (short)0, acc, false, false);
    }

    // ---- store Wh tile in swizzled bf16 B-fragment layout (1 x b128/lane) ----
    {
        const int b   = rt >> 6;
        const int jt  = (rt >> 1) & 31;
        const int qrt = rt & 1;
        unsigned int p0 = packbf(acc[0], acc[1]);
        unsigned int p1 = packbf(acc[2], acc[3]);
        unsigned int p2 = packbf(acc[4], acc[5]);
        unsigned int p3 = packbf(acc[6], acc[7]);
        v4u st = {p0, p1, p2, p3};
        size_t idx = ((size_t)((b * 32 + jt) * 16 + otile)) * 512
                   + (size_t)(16 * qrt + col) * 16 + (size_t)kh * 8;
        *(v4u*)(WhB + idx) = st;
    }

    // ---- ei/ej partials: reduce over this wave's 16 columns ----
    {
        const float a1 = avec[otile * 16 + col];
        const float a2 = avec[DD + otile * 16 + col];
        float pei[8], pej[8];
        #pragma unroll
        for (int v = 0; v < 8; ++v) { pei[v] = acc[v] * a1; pej[v] = acc[v] * a2; }
        #pragma unroll
        for (int m = 1; m <= 8; m <<= 1) {
            #pragma unroll
            for (int v = 0; v < 8; ++v) {
                pei[v] += __shfl_xor(pei[v], m, 16);
                pej[v] += __shfl_xor(pej[v], m, 16);
            }
        }
        if (col == 0) {
            #pragma unroll
            for (int v = 0; v < 8; ++v) {
                sEi[otile][v + 8 * kh] = pei[v];
                sEj[otile][v + 8 * kh] = pej[v];
            }
        }
    }
    __syncthreads();

    if (tid < 16) {
        float se = 0.f, sj = 0.f;
        #pragma unroll
        for (int w = 0; w < 16; ++w) { se += sEi[w][tid]; sj += sEj[w][tid]; }
        ei[r0 + tid] = se;
        ej[r0 + tid] = sj;
    }
}

// =====================================================================
// Kernel 3: fused attention for one (b, 16-row tile):
//   e -> masked leakyrelu -> row softmax (in LDS) -> alpha(bf16, A-layout)
//   -> h_out = alpha @ Wh   via 32x WMMA per wave (wave = 16-col otile)
// grid = 1024, block = 512, dynamic LDS ~100KB
// ej staging is done by the Tensor Data Mover (TDM) when available:
// 1D tile (1024 x f32 = 4KB), issued once by wave 0, s_wait_tensorcnt 0.
// =====================================================================
__global__ __launch_bounds__(512) void k_attn(const float* __restrict__ ei,
                                              const float* __restrict__ ej,
                                              const unsigned short* __restrict__ WhB,
                                              const int* __restrict__ adj,
                                              float* __restrict__ out) {
    extern __shared__ char smem[];
    float*        e_s    = (float*)smem;                          // 16*1024 f32 = 64KB
    unsigned int* sAlpha = (unsigned int*)(smem + 65536);         // 8192 u32 = 32KB
    float*        ej_s   = (float*)(smem + 65536 + 32768);        // 1024 f32
    float*        ei_s   = ej_s + 1024;                           // 16
    float*        row_m  = ei_s + 16;                             // 16
    float*        row_rs = row_m + 16;                            // 16

    const int blk  = blockIdx.x;
    const int b    = blk >> 6;
    const int n0   = (blk & 63) * 16;
    const int tid  = threadIdx.x;
    const int wave = tid >> 5;
    const int L    = tid & 31;
    const int col  = L & 15;
    const int kh   = L >> 4;

    // ---- stage ej for this batch via TDM (4KB 1D tile: global -> LDS) ----
#if __has_builtin(__builtin_amdgcn_tensor_load_to_lds) && __has_builtin(__builtin_amdgcn_s_wait_tensorcnt)
    if (wave == 0) {
        const unsigned long long ga =
            (unsigned long long)(uintptr_t)(ej + (size_t)b * NN);
        const unsigned int lds_a = (unsigned int)(uintptr_t)(void*)ej_s;
        // D# group0: count=1, lds_addr, global_addr[56:0], type=2 (bits 127:126)
        v4u g0 = { 1u,
                   lds_a,
                   (unsigned int)ga,
                   (unsigned int)(((ga >> 32) & 0x01FFFFFFu) | 0x80000000u) };
        // D# group1: data_size=4B (code 2), tensor_dim0=1024, tensor_dim1=1,
        //            tile_dim0=1024, tile_dim1=1, tensor_dim0_stride=1024
        v8i g1 = { (int)(2u << 16),          // [17:16] data_size
                   (int)(1024u << 16),       // [79:48] tensor_dim0 (low 16)
                   (int)(1u << 16),          // tensor_dim0 hi=0 | tensor_dim1 lo=1
                   (int)(1024u << 16),       // tensor_dim1 hi=0 | [127:112] tile_dim0
                   (int)1,                   // [143:128] tile_dim1=1, tile_dim2=0
                   (int)1024,                // [207:160] tensor_dim0_stride (low 32)
                   0, 0 };
        v4i g2 = { 1, 1, 0, 0 };             // tensor_dim2=1, tensor_dim3=1
        v4i g3 = { 0, 0, 0, 0 };
#if defined(__clang_major__) && __clang_major__ >= 23
        v8i g4 = { 0, 0, 0, 0, 0, 0, 0, 0 };
        __builtin_amdgcn_tensor_load_to_lds(g0, g1, g2, g3, g4, 0);
#else
        __builtin_amdgcn_tensor_load_to_lds(g0, g1, g2, g3, 0);
#endif
        __builtin_amdgcn_s_wait_tensorcnt(0);   // s_wait_tensorcnt 0
    }
#else
    for (int idx = tid; idx < NN; idx += 512) ej_s[idx] = ej[b * NN + idx];
#endif
    if (tid < 16) ei_s[tid] = ei[b * NN + n0 + tid];
    __syncthreads();

    // ---- e = mask(leakyrelu(ei + ej)) into LDS, 4 octets of 8 per thread ----
    #pragma unroll
    for (int s = 0; s < 4; ++s) {
        const int oct = tid + s * 512;       // 0..2047
        const int i   = oct >> 7;
        const int j0  = (oct & 127) * 8;
        const float bei = ei_s[i];
        const int* ap = adj + ((size_t)(b * NN + n0 + i)) * NN + j0;
        int4 m0 = *(const int4*)(ap);
        int4 m1 = *(const int4*)(ap + 4);
        float4 r0, r1;
        {
            float t;
            t = bei + ej_s[j0 + 0]; t = t > 0.f ? t : NEG_SLOPE * t; r0.x = m0.x ? t : MASKVAL;
            t = bei + ej_s[j0 + 1]; t = t > 0.f ? t : NEG_SLOPE * t; r0.y = m0.y ? t : MASKVAL;
            t = bei + ej_s[j0 + 2]; t = t > 0.f ? t : NEG_SLOPE * t; r0.z = m0.z ? t : MASKVAL;
            t = bei + ej_s[j0 + 3]; t = t > 0.f ? t : NEG_SLOPE * t; r0.w = m0.w ? t : MASKVAL;
            t = bei + ej_s[j0 + 4]; t = t > 0.f ? t : NEG_SLOPE * t; r1.x = m1.x ? t : MASKVAL;
            t = bei + ej_s[j0 + 5]; t = t > 0.f ? t : NEG_SLOPE * t; r1.y = m1.y ? t : MASKVAL;
            t = bei + ej_s[j0 + 6]; t = t > 0.f ? t : NEG_SLOPE * t; r1.z = m1.z ? t : MASKVAL;
            t = bei + ej_s[j0 + 7]; t = t > 0.f ? t : NEG_SLOPE * t; r1.w = m1.w ? t : MASKVAL;
        }
        *(float4*)(e_s + i * NN + j0)     = r0;
        *(float4*)(e_s + i * NN + j0 + 4) = r1;
    }
    __syncthreads();

    // ---- per-row max & sum-of-exp: wave w owns row w ----
    {
        float mx = -3.4e38f;
        for (int j = L; j < NN; j += 32) mx = fmaxf(mx, e_s[wave * NN + j]);
        #pragma unroll
        for (int d = 16; d >= 1; d >>= 1) mx = fmaxf(mx, __shfl_xor(mx, d, 32));
        float sum = 0.f;
        for (int j = L; j < NN; j += 32) sum += __expf(e_s[wave * NN + j] - mx);
        #pragma unroll
        for (int d = 16; d >= 1; d >>= 1) sum += __shfl_xor(sum, d, 32);
        if (L == 0) {
            row_m[wave]  = mx;
            row_rs[wave] = (mx < -9.0e29f) ? 0.f : (1.0f / sum);  // all-masked row -> 0
        }
    }
    __syncthreads();

    // ---- alpha (bf16) written directly in A-fragment layout in LDS ----
    #pragma unroll
    for (int s = 0; s < 4; ++s) {
        const int oct = tid + s * 512;
        const int i   = oct >> 7;
        const int c   = oct & 127;
        const int j0  = c * 8;
        const float mx = row_m[i];
        const float rs = row_rs[i];
        float4 x0 = *(const float4*)(e_s + i * NN + j0);
        float4 x1 = *(const float4*)(e_s + i * NN + j0 + 4);
        unsigned int p0 = packbf(__expf(x0.x - mx) * rs, __expf(x0.y - mx) * rs);
        unsigned int p1 = packbf(__expf(x0.z - mx) * rs, __expf(x0.w - mx) * rs);
        unsigned int p2 = packbf(__expf(x1.x - mx) * rs, __expf(x1.y - mx) * rs);
        unsigned int p3 = packbf(__expf(x1.z - mx) * rs, __expf(x1.w - mx) * rs);
        const int jt  = c >> 2;
        const int kk0 = (c & 3) * 8;
        const int q   = (kk0 >> 4) & 1;
        const int v0  = ((kk0 >> 3) & 1) * 4;
        v4u st = {p0, p1, p2, p3};
        ((v4u*)sAlpha)[(jt * 256 + (i + 16 * q) * 8 + v0) >> 2] = st;
    }
    __syncthreads();

    // ---- h_out tile: 32x WMMA over j, wave = otile ----
    const int otile = wave;
    v8f acc = {};
    #pragma unroll 4
    for (int jt = 0; jt < 32; ++jt) {
        Frag16 fa, fb;
        const v4u* ap = (const v4u*)(sAlpha + jt * 256 + L * 8);
        fa.u[0] = ap[0];
        fa.u[1] = ap[1];
        const unsigned short* bbase =
            WhB + ((size_t)((b * 32 + jt) * 16 + otile)) * 512 + (size_t)L * 16;
        const v4u* bp = (const v4u*)bbase;
        fb.u[0] = bp[0];
        fb.u[1] = bp[1];
        if (jt + 1 < 32)   // -> global_prefetch_b8
            __builtin_prefetch(bbase + 16 * 512, 0, 0);
        acc = __builtin_amdgcn_wmma_f32_16x16x32_bf16(
            false, fa.b, false, fb.b, (short)0, acc, false, false);
    }
    #pragma unroll
    for (int v = 0; v < 8; ++v) {
        const int m = v + 8 * kh;
        out[((size_t)(b * NN + n0 + m)) * DD + otile * 16 + col] = acc[v];
    }
}

// =====================================================================
extern "C" void kernel_launch(void* const* d_in, const int* in_sizes, int n_in,
                              void* d_out, int out_size, void* d_ws, size_t ws_size,
                              hipStream_t stream) {
    (void)in_sizes; (void)n_in; (void)out_size; (void)ws_size;
    const float* h    = (const float*)d_in[0];   // (16,1024,256) f32
    const int*   adj  = (const int*)d_in[1];     // (16,1024,1024) i32
    const float* W    = (const float*)d_in[2];   // (256,256) f32
    const float* avec = (const float*)d_in[3];   // (512,) f32
    float* out = (float*)d_out;                  // (16,1024,256) f32

    // workspace layout
    char* ws = (char*)d_ws;
    unsigned short* WhB  = (unsigned short*)(ws);                    // 8 MB
    float*          ei   = (float*)(ws + 8388608);                   // 64 KB
    float*          ej   = (float*)(ws + 8388608 + 65536);           // 64 KB
    unsigned short* WbfB = (unsigned short*)(ws + 8388608 + 131072); // 128 KB

    k_wswz<<<dim3(256), dim3(256), 0, stream>>>(W, WbfB);
    k_wh<<<dim3(1024), dim3(512), 0, stream>>>(h, avec, WbfB, WhB, ei, ej);

    const size_t smem = 65536 + 32768 + 4096 + 64 + 64 + 64; // 102,592 B
    k_attn<<<dim3(1024), dim3(512), smem, stream>>>(ei, ej, WhB, adj, out);
}
